// ClusterAttention_26929444946663
// MI455X (gfx1250) — compile-verified, matrix operand order checked by reading, not run on hardware
//
#include <hip/hip_runtime.h>
#include <hip/hip_bf16.h>

// ---------------------------------------------------------------------------
// ClusterAttention on MI455X (gfx1250), wave32.
//  - All matmuls on v_wmma_f32_16x16x32_f16 (f32 accumulate).
//  - Global->LDS tile staging via GLOBAL_LOAD_ASYNC_TO_LDS_B128 (ASYNCcnt),
//    double-buffered so tile k+1 copies overlap WMMAs on tile k.
//  - Weights pre-transposed+converted to f16 so A and B tiles are identical
//    contiguous-along-K 16B async copies.
//  - Positional bias is separable: only +t[j] survives softmax invariance.
// ---------------------------------------------------------------------------

typedef __attribute__((ext_vector_type(16))) _Float16 v16h;
typedef __attribute__((ext_vector_type(8)))  _Float16 v8h;
typedef __attribute__((ext_vector_type(8)))  float    v8f;

#define CB 4
#define CN 8192
#define CC 512
#define CH 8
#define CK 64
#define CM 128
#define CCH 64          // c_ = C/H
#define C3C 1536        // 3*C

#define USE_ASYNC 1

__device__ __forceinline__ v8f wmma_f16(v16h a, v16h b, v8f c) {
    return __builtin_amdgcn_wmma_f32_16x16x32_f16(
        false, a, false, b, (short)0, c, false, false);
}

__device__ __forceinline__ v8f vzero8f() {
    v8f z = {0.f, 0.f, 0.f, 0.f, 0.f, 0.f, 0.f, 0.f};
    return z;
}

// 16-byte global -> LDS copy. Async variant issues GLOBAL_LOAD_ASYNC_TO_LDS_B128
// (per-lane: 16B from its global address to its LDS byte offset, ASYNCcnt).
__device__ __forceinline__ void cp16_g2l(void* ldst, const void* gsrc) {
#if USE_ASYNC
    unsigned loff = (unsigned)(size_t)ldst;            // AS3 byte offset
    unsigned long long ga = (unsigned long long)(size_t)gsrc;
    asm volatile("global_load_async_to_lds_b128 %0, %1, off"
                 :: "v"(loff), "v"(ga) : "memory");
#else
    *(uint4*)ldst = *(const uint4*)gsrc;
#endif
}
__device__ __forceinline__ void async_wait0() {
#if USE_ASYNC
    asm volatile("s_wait_asynccnt 0x0" ::: "memory");
#endif
}
__device__ __forceinline__ void async_wait4() {
#if USE_ASYNC
    asm volatile("s_wait_asynccnt 0x4" ::: "memory");
#endif
}

// 16x32 f16 fragment load (A layout; B identical with column-major storage).
// Lane L<16: row L, K {0..7}+{16..23}; lane L+16: row L, K {8..15}+{24..31}.
__device__ __forceinline__ v16h frag_ld(const _Float16* p, int stride_h, int lane) {
    const _Float16* rp = p + (lane & 15) * stride_h + ((lane >> 4) << 3);
    v8h lo = *(const v8h*)(rp);
    v8h hi = *(const v8h*)(rp + 16);
    return __builtin_shufflevector(lo, hi, 0,1,2,3,4,5,6,7,8,9,10,11,12,13,14,15);
}

// ---------------------------------------------------------------------------
// pos max reduction (pos >= 0 -> uint-bit atomicMax is order-preserving)
// ---------------------------------------------------------------------------
__global__ void posmax_init(unsigned* pm) {
    if (threadIdx.x == 0) { pm[0] = 0u; pm[1] = 0u; }
}

__global__ __launch_bounds__(256) void posmax_kernel(const float* __restrict__ pos,
                                                     int npairs, unsigned* pm) {
    __shared__ float r0[256], r1[256];
    float m0 = 0.f, m1 = 0.f;
    for (int i = blockIdx.x * blockDim.x + threadIdx.x; i < npairs;
         i += gridDim.x * blockDim.x) {
        m0 = fmaxf(m0, pos[2 * i + 0]);
        m1 = fmaxf(m1, pos[2 * i + 1]);
    }
    r0[threadIdx.x] = m0; r1[threadIdx.x] = m1;
    __syncthreads();
    for (int s = 128; s > 0; s >>= 1) {
        if (threadIdx.x < s) {
            r0[threadIdx.x] = fmaxf(r0[threadIdx.x], r0[threadIdx.x + s]);
            r1[threadIdx.x] = fmaxf(r1[threadIdx.x], r1[threadIdx.x + s]);
        }
        __syncthreads();
    }
    if (threadIdx.x == 0) {
        atomicMax(&pm[0], __float_as_uint(r0[0]));
        atomicMax(&pm[1], __float_as_uint(r1[0]));
    }
}

// ---------------------------------------------------------------------------
// f32 -> f16 elementwise (vectorized by 4)
// ---------------------------------------------------------------------------
__global__ __launch_bounds__(256) void cvt_f16_kernel(const float* __restrict__ in,
                                                      _Float16* __restrict__ out,
                                                      int n4) {
    for (int i = blockIdx.x * blockDim.x + threadIdx.x; i < n4;
         i += gridDim.x * blockDim.x) {
        float4 f = ((const float4*)in)[i];
        _Float16* o = out + (size_t)i * 4;
        o[0] = (_Float16)f.x; o[1] = (_Float16)f.y;
        o[2] = (_Float16)f.z; o[3] = (_Float16)f.w;
    }
}

// W (Kdim x Ndim f32, row-major) -> WT (Ndim x Kdim f16, row-major)
__global__ __launch_bounds__(256) void wt_kernel(const float* __restrict__ W,
                                                 _Float16* __restrict__ WT,
                                                 int Kdim, int Ndim) {
    int total = Kdim * Ndim;
    for (int i = blockIdx.x * blockDim.x + threadIdx.x; i < total;
         i += gridDim.x * blockDim.x) {
        int n = i / Kdim, k = i - n * Kdim;
        WT[i] = (_Float16)W[(size_t)k * Ndim + n];
    }
}

// ---------------------------------------------------------------------------
// GEMM: Out(M x Ndim) = A(M x Kdim f16) * WT^T + bias ; WT is (Ndim x Kdim f16).
// Block tile 128x128, K-step 32, 8 waves 4x2, 32x64 per wave (8 WMMA/step).
// Double-buffered LDS, async global->LDS copies.
// ---------------------------------------------------------------------------
#define GS 40   // LDS row stride in halves (80B, multiple of 16B)

template <bool OUT_HALF>
__global__ __launch_bounds__(256) void gemm_tt(const _Float16* __restrict__ A,
                                               const _Float16* __restrict__ WT,
                                               const float* __restrict__ bias,
                                               void* __restrict__ Outp,
                                               int Kdim, int Ndim) {
    __shared__ _Float16 sA[2][128 * GS];
    __shared__ _Float16 sB[2][128 * GS];

    const int tid  = threadIdx.x;
    const int lane = tid & 31;
    const int wave = tid >> 5;
    const int mBase = blockIdx.x * 128;
    const int nBase = blockIdx.y * 128;
    const int wm = (wave & 3) * 32;   // wave rows
    const int wn = (wave >> 2) * 64;  // wave cols

    v8f acc[2][4];
    #pragma unroll
    for (int tm = 0; tm < 2; ++tm)
        #pragma unroll
        for (int tn = 0; tn < 4; ++tn) acc[tm][tn] = vzero8f();

    // Per-tile staging: A tile 128x32 and B tile 128x32, each 512 chunks of
    // 8 halves; per thread 2 chunks of each => 4 async ops per tile.
    const int nk = Kdim >> 5;
    auto issue_tile = [&](int buf, int k0) {
        #pragma unroll
        for (int it = 0; it < 2; ++it) {
            int lin = tid + it * 256;            // 0..511
            int row = lin >> 2, c8 = (lin & 3) << 3;
            cp16_g2l(&sA[buf][row * GS + c8],
                     A  + (size_t)(mBase + row) * Kdim + k0 + c8);
            cp16_g2l(&sB[buf][row * GS + c8],
                     WT + (size_t)(nBase + row) * Kdim + k0 + c8);
        }
    };

    issue_tile(0, 0);
    for (int k = 0; k < nk; ++k) {
        const int cur = k & 1;
        if (k + 1 < nk) {
            issue_tile(1 - cur, (k + 1) << 5);
            async_wait4();                 // tile k landed; tile k+1 in flight
        } else {
            async_wait0();
        }
        __syncthreads();

        v16h a0 = frag_ld(&sA[cur][(wm + 0)  * GS], GS, lane);
        v16h a1 = frag_ld(&sA[cur][(wm + 16) * GS], GS, lane);
        #pragma unroll
        for (int tn = 0; tn < 4; ++tn) {
            v16h bf = frag_ld(&sB[cur][(wn + tn * 16) * GS], GS, lane);
            acc[0][tn] = wmma_f16(a0, bf, acc[0][tn]);
            acc[1][tn] = wmma_f16(a1, bf, acc[1][tn]);
        }
        __syncthreads();
    }

    const int rb = (lane >> 4) << 3;
    const int cl = lane & 15;
    #pragma unroll
    for (int tm = 0; tm < 2; ++tm)
        #pragma unroll
        for (int tn = 0; tn < 4; ++tn)
            #pragma unroll
            for (int r = 0; r < 8; ++r) {
                int row = mBase + wm + tm * 16 + rb + r;
                int col = nBase + wn + tn * 16 + cl;
                float v = acc[tm][tn][r] + bias[col];
                if (OUT_HALF)
                    ((_Float16*)Outp)[(size_t)row * Ndim + col] = (_Float16)v;
                else
                    ((float*)Outp)[(size_t)row * Ndim + col] = v;
            }
}

// ---------------------------------------------------------------------------
// Per-cluster attention: one block per (b,k,h). 256 threads = 8 waves.
// ---------------------------------------------------------------------------
#define STQ 72    // q/k LDS row stride (halves): 144B
#define STP 136   // S/P and V^T LDS row stride (halves): 272B

__global__ __launch_bounds__(256) void attn_kernel(
    const _Float16* __restrict__ qkv,       // (B*N,1536): h*192 + {q,k,v}*64 + c
    const int* __restrict__ member_idx,     // (B,H,K,M)
    const float* __restrict__ cluster_mask, // (B*H,K,M)
    const float* __restrict__ pos,          // (B,N,2)
    const float* __restrict__ w_pos,        // (H,2)
    const unsigned* __restrict__ posmax,    // 2 x f32 bits
    _Float16* __restrict__ feat_out)        // (B*N,512)
{
    // sreg: q(128xSTQ)|k(128xSTQ) in phase 1; aliased by S/P (128xSTP) after.
    __shared__ _Float16 sreg[2 * 128 * STQ];      // 18432 halves
    __shared__ _Float16 svt[CCH * STP];           // V^T: 64 x 136
    __shared__ float sT[CM];
    __shared__ float sMask[CM];
    __shared__ int   sIdx[CM];

    _Float16* sq  = sreg;
    _Float16* sk  = sreg + 128 * STQ;
    _Float16* sSP = sreg;                          // 128*STP = 17408 <= 18432

    const int tid  = threadIdx.x;
    const int lane = tid & 31;
    const int wave = tid >> 5;

    const int cid = blockIdx.x;
    const int h  = cid & (CH - 1);
    const int kk = (cid >> 3) & (CK - 1);
    const int b  = cid >> 9;

    // ---- phase 0: cluster metadata ----
    if (tid < CM) {
        size_t off = (((size_t)b * CH + h) * CK + kk) * CM + tid;
        int n = member_idx[off];
        sIdx[tid]  = n;
        sMask[tid] = cluster_mask[off];
        float inv0 = 1.0f / __uint_as_float(posmax[0]);
        float inv1 = 1.0f / __uint_as_float(posmax[1]);
        const float* pp = pos + ((size_t)b * CN + n) * 2;
        sT[tid] = pp[0] * inv0 * w_pos[h * 2 + 0] + pp[1] * inv1 * w_pos[h * 2 + 1];
    }
    __syncthreads();

    // ---- phase 1: gather q,k (async 16B copies) and v (transposed, scalar) --
    {
        int row = tid >> 1, hf = tid & 1;
        const _Float16* src = qkv + ((size_t)b * CN + sIdx[row]) * C3C + h * 192;
        const char* sb = (const char*)src;
        char* dqb = (char*)&sq[row * STQ];
        char* dkb = (char*)&sk[row * STQ];
        #pragma unroll
        for (int i = 0; i < 4; ++i) {
            int o = (hf * 4 + i) * 16;
            cp16_g2l(dqb + o, sb + o);            // q: halves [0,64)
            cp16_g2l(dkb + o, sb + 128 + o);      // k: halves [64,128)
        }
        const _Float16* vs = src + 128;           // v: halves [128,192)
        for (int c = hf * 32; c < hf * 32 + 32; ++c)
            svt[c * STP + row] = vs[c];
    }
    async_wait0();
    __syncthreads();

    // ---- phase 2: S = Q K^T (128x128), waves 4(M) x 2(N of 64) ----
    const int wmS = (wave & 3) * 32;
    const int wnS = (wave >> 2) * 64;
    v8f s[2][4];
    #pragma unroll
    for (int tm = 0; tm < 2; ++tm)
        #pragma unroll
        for (int tn = 0; tn < 4; ++tn) s[tm][tn] = vzero8f();

    #pragma unroll
    for (int ks = 0; ks < CCH; ks += 32) {
        v16h a0 = frag_ld(&sq[(wmS + 0)  * STQ + ks], STQ, lane);
        v16h a1 = frag_ld(&sq[(wmS + 16) * STQ + ks], STQ, lane);
        #pragma unroll
        for (int tn = 0; tn < 4; ++tn) {
            v16h bf = frag_ld(&sk[(wnS + tn * 16) * STQ + ks], STQ, lane);
            s[0][tn] = wmma_f16(a0, bf, s[0][tn]);
            s[1][tn] = wmma_f16(a1, bf, s[1][tn]);
        }
    }
    __syncthreads();   // all q/k reads done before aliasing writes

    const float scale = 0.125f;    // 64^-0.5
    const int rb = (lane >> 4) << 3;
    const int cl = lane & 15;
    #pragma unroll
    for (int tm = 0; tm < 2; ++tm)
        #pragma unroll
        for (int tn = 0; tn < 4; ++tn)
            #pragma unroll
            for (int r = 0; r < 8; ++r) {
                int row = wmS + tm * 16 + rb + r;
                int col = wnS + tn * 16 + cl;
                sSP[row * STP + col] = (_Float16)(s[tm][tn][r] * scale);
            }
    __syncthreads();

    // ---- phase 3: row softmax with separable bias +t[j] and mask ----
    if (tid < CM) {
        _Float16* rowp = &sSP[tid * STP];
        float mx = -1e30f;
        for (int j = 0; j < CM; ++j) {
            float x = (float)rowp[j] + sT[j] + (1.0f - sMask[j]) * -100.0f;
            rowp[j] = (_Float16)x;
            mx = fmaxf(mx, x);
        }
        float sum = 0.f;
        for (int j = 0; j < CM; ++j) {
            float e = expf((float)rowp[j] - mx);
            rowp[j] = (_Float16)e;
            sum += e;
        }
        float inv = 1.0f / sum;
        for (int j = 0; j < CM; ++j)
            rowp[j] = (_Float16)((float)rowp[j] * inv);
    }
    __syncthreads();

    // ---- phase 4: out = P (128x128) @ V (128x64), waves 4(M) x 2(N of 32) --
    const int wmO = (wave & 3) * 32;
    const int wnO = (wave >> 2) * 32;
    v8f o00 = vzero8f(), o01 = vzero8f(), o10 = vzero8f(), o11 = vzero8f();
    #pragma unroll
    for (int ks = 0; ks < CM; ks += 32) {
        v16h a0 = frag_ld(&sSP[(wmO + 0)  * STP + ks], STP, lane);
        v16h a1 = frag_ld(&sSP[(wmO + 16) * STP + ks], STP, lane);
        v16h b0 = frag_ld(&svt[(wnO + 0)  * STP + ks], STP, lane);
        v16h b1 = frag_ld(&svt[(wnO + 16) * STP + ks], STP, lane);
        o00 = wmma_f16(a0, b0, o00);
        o01 = wmma_f16(a0, b1, o01);
        o10 = wmma_f16(a1, b0, o10);
        o11 = wmma_f16(a1, b1, o11);
    }

    // ---- phase 5: scale by mask[i], scatter rows to feat_out ----
    v8f outs[4] = {o00, o01, o10, o11};
    #pragma unroll
    for (int t = 0; t < 4; ++t) {
        int tm = t >> 1, tn = t & 1;
        #pragma unroll
        for (int r = 0; r < 8; ++r) {
            int i = wmO + tm * 16 + rb + r;      // member row
            int c = wnO + tn * 16 + cl;          // channel
            float v = outs[t][r] * sMask[i];
            feat_out[((size_t)b * CN + sIdx[i]) * CC + h * CCH + c] = (_Float16)v;
        }
    }
}

// ---------------------------------------------------------------------------
// Launch
// ---------------------------------------------------------------------------
extern "C" void kernel_launch(void* const* d_in, const int* in_sizes, int n_in,
                              void* d_out, int out_size, void* d_ws, size_t ws_size,
                              hipStream_t stream) {
    const float* pos        = (const float*)d_in[0];
    const float* feat       = (const float*)d_in[1];
    const int*   member_idx = (const int*)d_in[2];
    const float* cmask      = (const float*)d_in[3];
    const float* w_qkv      = (const float*)d_in[4];
    const float* b_qkv      = (const float*)d_in[5];
    const float* w_pos      = (const float*)d_in[6];
    // d_in[7] = b_pos: cancels in softmax (constant per row)
    const float* w_proj     = (const float*)d_in[8];
    const float* b_proj     = (const float*)d_in[9];

    // Workspace: qkv f16 | feat_out f16 | feat f16 | w_qkv^T f16 | w_proj^T f16 | posmax
    _Float16* qkv_h  = (_Float16*)d_ws;
    _Float16* fo_h   = qkv_h  + (size_t)CB * CN * C3C;
    _Float16* feat_h = fo_h   + (size_t)CB * CN * CC;
    _Float16* wqkvT  = feat_h + (size_t)CB * CN * CC;
    _Float16* wprojT = wqkvT  + (size_t)CC * C3C;
    unsigned* pm     = (unsigned*)(wprojT + (size_t)CC * CC);

    posmax_init<<<1, 64, 0, stream>>>(pm);
    posmax_kernel<<<64, 256, 0, stream>>>(pos, CB * CN, pm);

    cvt_f16_kernel<<<2048, 256, 0, stream>>>(feat, feat_h, CB * CN * CC / 4);
    wt_kernel<<<512, 256, 0, stream>>>(w_qkv, wqkvT, CC, C3C);
    wt_kernel<<<256, 256, 0, stream>>>(w_proj, wprojT, CC, CC);

    // qkv = feat @ w_qkv + b_qkv -> f16
    gemm_tt<true><<<dim3((CB * CN) / 128, C3C / 128), 256, 0, stream>>>(
        feat_h, wqkvT, b_qkv, qkv_h, CC, C3C);

    // per-cluster attention
    attn_kernel<<<CB * CK * CH, 256, 0, stream>>>(
        qkv_h, member_idx, cmask, pos, w_pos, pm, fo_h);

    // out = feat_out @ w_proj + b_proj -> f32
    gemm_tt<false><<<dim3((CB * CN) / 128, CC / 128), 256, 0, stream>>>(
        fo_h, wprojT, b_proj, d_out, CC, CC);
}
